// PoolingLayer_57561151701571
// MI455X (gfx1250) — matrix operand order repaired and checked by reading, not compiled, not run on hardware
//
#include <hip/hip_runtime.h>
#include <math.h>

// ---------------- problem constants (fixed by the reference) ----------------
#define N_TOTAL   262144
#define C_DIM     128
#define E_EDGES   4194304
#define ED_DIM    8
#define NG        64
#define NPER      4096          // nodes per graph
#define KSEL      2048          // ceil(0.5 * 4096)
#define POOLED    (NG * KSEL)   // 131072 pooled nodes

typedef __attribute__((ext_vector_type(16))) _Float16 v16h;
typedef __attribute__((ext_vector_type(8)))  float    v8f;

// ---------------------------------------------------------------------------
// Kernel 1: raw projection scores via WMMA.  One wave = 16 rows, K=128 as
// four chained v_wmma_f32_16x16x32_f16.  B holds w broadcast into all 16
// columns, so every D column equals the score; we read column N=0.
// Top-k ordering is invariant to the positive 1/||w|| scale, so raw scores
// suffice here; the norm is folded into tanh later.
// ---------------------------------------------------------------------------
__global__ __launch_bounds__(256) void score_wmma_kernel(
    const float* __restrict__ x, const float* __restrict__ w,
    float* __restrict__ scores) {
  const int lane = threadIdx.x & 31;
  const int wave = threadIdx.x >> 5;
  const int tile = blockIdx.x * (blockDim.x >> 5) + wave;   // 16 rows per tile
  const int row0 = tile * 16;
  const int m    = lane & 15;
  const int hi   = lane >> 4;                                // lane half-group

  const float* xrow = x + (size_t)(row0 + m) * C_DIM;

  v8f acc = {};
#pragma unroll
  for (int kb = 0; kb < C_DIM; kb += 32) {
    // A fragment (16x32 f16): lanes 0-15: elems 0-7 = K kb..kb+7, 8-15 = K kb+16..kb+23
    //                         lanes 16-31: shift K base by +8.
    v16h a, b;
    const int ka0 = kb + hi * 8;
    const int ka1 = kb + 16 + hi * 8;
#pragma unroll
    for (int j = 0; j < 8; ++j) {
      a[j]     = (_Float16)xrow[ka0 + j];
      a[8 + j] = (_Float16)xrow[ka1 + j];
    }
    // B fragment (32x16 f16): same K order as A (elems 0-7 = K ka0.., 8-15 = K ka1..);
    // broadcast w so every column is identical.
#pragma unroll
    for (int j = 0; j < 8; ++j) {
      b[j]     = (_Float16)w[ka0 + j];
      b[8 + j] = (_Float16)w[ka1 + j];
    }

    acc = __builtin_amdgcn_wmma_f32_16x16x32_f16(
        /*neg_a=*/false, a, /*neg_b=*/false, b,
        /*c_mod=*/(short)0, acc, /*reuse_a=*/false, /*reuse_b=*/false);
  }
  // D layout: VGPR r -> (M=r, N=lane) for lanes 0-15, (M=r+8, N=lane-16) for 16-31.
  if (lane == 0) {
#pragma unroll
    for (int r = 0; r < 8; ++r) scores[row0 + r] = acc[r];
  } else if (lane == 16) {
#pragma unroll
    for (int r = 0; r < 8; ++r) scores[row0 + 8 + r] = acc[r];
  }
}

// ---------------------------------------------------------------------------
// Kernel 0: node_new := -1
// ---------------------------------------------------------------------------
__global__ void init_nodemap_kernel(int* __restrict__ node_new) {
  int i = blockIdx.x * blockDim.x + threadIdx.x;
  node_new[i] = -1;
}

// ---------------------------------------------------------------------------
// Kernel 2: per-graph bitonic top-k in LDS.  Key = (~asc_map(score) << 32) | idx
// -> ascending u64 sort gives descending score, ties broken by smaller index
// (matches jax.lax.top_k).  Emits perm/batch outputs, tanh gains, gather
// indices, and scatters the old->new node map.
// ---------------------------------------------------------------------------
__global__ __launch_bounds__(1024) void topk_kernel(
    const float* __restrict__ scores, const float* __restrict__ w,
    int* __restrict__ node_new, int* __restrict__ perm_idx,
    float* __restrict__ gain, float* __restrict__ out_perm,
    float* __restrict__ out_batch) {
  __shared__ unsigned long long keys[NPER];
  __shared__ float s_red[C_DIM];
  __shared__ float s_invnorm;

  const int g = blockIdx.x;
  const int t = threadIdx.x;
  const float* sg = scores + (size_t)g * NPER;

  // redundant per-block 1/||w|| (128 floats, trivial)
  if (t < C_DIM) { float v = w[t]; s_red[t] = v * v; }
  __syncthreads();
  if (t == 0) {
    float s = 0.f;
#pragma unroll
    for (int i = 0; i < C_DIM; ++i) s += s_red[i];
    s_invnorm = rsqrtf(s);
  }

  // build sort keys
  for (int i = t; i < NPER; i += 1024) {
    unsigned u = __float_as_uint(sg[i]);
    u = (u & 0x80000000u) ? ~u : (u | 0x80000000u);   // monotonic ascending map
    keys[i] = ((unsigned long long)(~u) << 32) | (unsigned)i;
  }
  __syncthreads();

  // bitonic sort, ascending
  for (int kk = 2; kk <= NPER; kk <<= 1) {
    for (int j = kk >> 1; j > 0; j >>= 1) {
      for (int i = t; i < NPER; i += 1024) {
        int ixj = i ^ j;
        if (ixj > i) {
          bool up = ((i & kk) == 0);
          unsigned long long A = keys[i], B = keys[ixj];
          if ((A > B) == up) { keys[i] = B; keys[ixj] = A; }
        }
      }
      __syncthreads();
    }
  }

  const float invn = s_invnorm;
  for (int jj = t; jj < KSEL; jj += 1024) {
    int idx   = (int)(keys[jj] & 0xFFFFFFFFull);
    int gidx  = g * NPER + idx;       // global old node id
    int jglob = g * KSEL + jj;        // new node id (position in perm)
    perm_idx[jglob]  = gidx;
    gain[jglob]      = tanhf(sg[idx] * invn);
    node_new[gidx]   = jglob;
    out_perm[jglob]  = (float)gidx;
    out_batch[jglob] = (float)g;
  }
}

// ---------------------------------------------------------------------------
// Kernel 3: x_pool[j] = x[perm[j]] * gain[j]   (float4 gather, BW-bound)
// ---------------------------------------------------------------------------
__global__ __launch_bounds__(256) void pool_kernel(
    const float* __restrict__ x, const int* __restrict__ perm_idx,
    const float* __restrict__ gain, float* __restrict__ xout) {
  int tid = blockIdx.x * blockDim.x + threadIdx.x;   // one thread per float4
  int row = tid >> 5;
  int q   = tid & 31;
  int src = perm_idx[row];
  float gn = gain[row];
  const float4* xs = (const float4*)(x + (size_t)src * C_DIM);
  float4 v = xs[q];
  v.x *= gn; v.y *= gn; v.z *= gn; v.w *= gn;
  ((float4*)(xout + (size_t)row * C_DIM))[q] = v;
}

// ---------------------------------------------------------------------------
// Kernel 4: edge filtering.  One thread per edge.
// ---------------------------------------------------------------------------
__global__ __launch_bounds__(256) void edge_kernel(
    const int* __restrict__ ei, const float* __restrict__ ea,
    const int* __restrict__ node_new, float* __restrict__ out_ei,
    float* __restrict__ out_ea, float* __restrict__ out_mask) {
  int e = blockIdx.x * blockDim.x + threadIdx.x;
  __builtin_prefetch(ea + (size_t)e * ED_DIM, 0, 0);   // global_prefetch_b8

  int src = ei[e]           & (N_TOTAL - 1);   // mask only as memory-safety
  int dst = ei[E_EDGES + e] & (N_TOTAL - 1);
  int sn = node_new[src];
  int dn = node_new[dst];
  bool m = (sn >= 0) && (dn >= 0);

  out_ei[e]           = m ? (float)sn : -1.0f;
  out_ei[E_EDGES + e] = m ? (float)dn : -1.0f;
  out_mask[e]         = m ? 1.0f : 0.0f;

  const float4* a = (const float4*)(ea + (size_t)e * ED_DIM);
  float4 a0 = a[0], a1 = a[1];
  if (!m) { a0 = make_float4(0, 0, 0, 0); a1 = make_float4(0, 0, 0, 0); }
  float4* o = (float4*)(out_ea + (size_t)e * ED_DIM);
  o[0] = a0; o[1] = a1;
}

// ---------------------------------------------------------------------------
extern "C" void kernel_launch(void* const* d_in, const int* in_sizes, int n_in,
                              void* d_out, int out_size, void* d_ws, size_t ws_size,
                              hipStream_t stream) {
  const float* x  = (const float*)d_in[0];
  const int*   ei = (const int*)d_in[1];
  const float* ea = (const float*)d_in[2];
  const float* w  = (const float*)d_in[3];
  // d_in[4] = num_graphs (compile-time constant NG here)

  // workspace layout (≈3 MB)
  float* scores   = (float*)d_ws;
  int*   node_new = (int*)((char*)d_ws + (size_t)N_TOTAL * 4);
  int*   perm_idx = (int*)((char*)d_ws + (size_t)N_TOTAL * 8);
  float* gain     = (float*)((char*)d_ws + (size_t)N_TOTAL * 8 + (size_t)POOLED * 4);

  // output layout: concatenated reference tuple, all as float
  float* out       = (float*)d_out;
  float* out_xpool = out;                                       // POOLED*C
  float* out_ei    = out_xpool + (size_t)POOLED * C_DIM;        // 2*E
  float* out_ea    = out_ei + 2 * (size_t)E_EDGES;              // E*ED
  float* out_batch = out_ea + (size_t)E_EDGES * ED_DIM;         // POOLED
  float* out_perm  = out_batch + POOLED;                        // POOLED
  float* out_mask  = out_perm + POOLED;                         // E

  init_nodemap_kernel<<<N_TOTAL / 256, 256, 0, stream>>>(node_new);

  // 16384 tiles of 16 rows, 8 waves/block -> 2048 blocks
  score_wmma_kernel<<<(N_TOTAL / 16) / 8, 256, 0, stream>>>(x, w, scores);

  topk_kernel<<<NG, 1024, 0, stream>>>(scores, w, node_new, perm_idx, gain,
                                       out_perm, out_batch);

  // POOLED rows * 32 float4 per row = 4,194,304 threads
  pool_kernel<<<(POOLED * 32) / 256, 256, 0, stream>>>(x, perm_idx, gain,
                                                       out_xpool);

  edge_kernel<<<E_EDGES / 256, 256, 0, stream>>>(ei, ea, node_new, out_ei,
                                                 out_ea, out_mask);
}
